// SPDNet3BiRe_77412490543692
// MI455X (gfx1250) — matrix operand order repaired and checked by reading, not compiled, not run on hardware
//
#include <hip/hip_runtime.h>
#include <hip/hip_bf16.h>

typedef __attribute__((ext_vector_type(2)))  float  v2f;
typedef __attribute__((ext_vector_type(8)))  float  v8f;
typedef __attribute__((ext_vector_type(16))) __bf16 v16bf;

#define NTHREADS 256
#define NWAVES   8
#define SPD_EPS  0.01f

// f32 LDS strides (pad 4: 16-lane column walks avoid single-bank)
#define LD64 68
#define LD32 36
#define LD16 20

// bf16 epilogue modes for gemm_bf
#define EP_F32  0   // Cf[r][c]  = alpha*acc            (f32)
#define EP_BF   1   // Cb[r][c]  = bf16(alpha*acc)
#define EP_BF_T 2   // Cb[c][r]  = bf16(alpha*acc)      (store transposed)
#define EP_BF_H 3   // Cb[r][c]  = bf16(3I - acc)       (Newton-Schulz H)

// Hardware bf16 converts (clang lowers fptrunc/fpext for __bf16 on gfx1250).
__device__ __forceinline__ __bf16 to_bf(float f)   { return (__bf16)f; }
__device__ __forceinline__ float  from_bf(__bf16 b){ return (float)b;  }

// ---------------------------------------------------------------------------
// f32 GEMM tiles via V_WMMA_F32_16X16X4_F32 (used for the 16x16 stages).
// C[M,N] = alpha * op(A)*B. `wave` is readfirstlane'd by the caller so the
// tile loop is scalar-uniform: EXEC stays all-ones around every WMMA.
// ---------------------------------------------------------------------------
__device__ __forceinline__ void gemm_tiles(
    const float* __restrict__ A, int lda, int transA,
    const float* __restrict__ B, int ldb,
    float* __restrict__ C, int ldc,
    int M, int N, int K, float alpha, int wave, int lane)
{
  __syncthreads();
  const int tilesN = N >> 4;
  const int ntiles = (M >> 4) * tilesN;
  const int l    = lane & 15;
  const int koff = (lane >= 16) ? 2 : 0;   // A/B lanes 16-31 hold K+2,K+3
  for (int t = wave; t < ntiles; t += NWAVES) {
    const int row0 = (t / tilesN) << 4;
    const int col0 = (t % tilesN) << 4;
    const int aBase = transA ? (row0 + l) : (row0 + l) * lda;
    const int aK    = transA ? lda : 1;
    const int bBase = col0 + l;
    v8f acc = {};
    for (int k0 = 0; k0 < K; k0 += 4) {
      const int ka = k0 + koff;
      v2f a, b;
      a.x = A[aBase + ka * aK];
      a.y = A[aBase + (ka + 1) * aK];
      b.x = B[ka * ldb + bBase];
      b.y = B[(ka + 1) * ldb + bBase];
      acc = __builtin_amdgcn_wmma_f32_16x16x4_f32(false, a, false, b,
                                                  (short)0, acc, false, false);
    }
    const int ro = (lane >= 16) ? 8 : 0;
#pragma unroll
    for (int r = 0; r < 8; ++r)
      C[(row0 + r + ro) * ldc + col0 + l] = alpha * acc[r];
  }
  __syncthreads();
}

// ---------------------------------------------------------------------------
// bf16 GEMM tiles via V_WMMA_F32_16X16X32_BF16 (f32 accumulate).
//   A : M x K bf16, row-major, lda elements (lda % 8 == 0)
//   BT: N x K bf16, row-major, ldb elements  -- i.e. B[k][n] = BT[n][k].
//       (For symmetric operands pass the matrix itself as BT.)
// ISA 16-bit layouts:
//   A lane(l,hi): elems 0-7 = K(k0+hi*8 .. +7), elems 8-15 = K(k0+16+hi*8 ..)
//   B lane(l,hi): elems 0-7 = K(k0+hi*8 .. +7), elems 8-15 = K(k0+16+hi*8 ..), col = l
// K must be a multiple of 32; all chunk offsets stay 16B aligned.
// ---------------------------------------------------------------------------
union BfPack { v16bf v; uint4 u[2]; };

__device__ __forceinline__ void gemm_bf(
    const __bf16* __restrict__ A, int lda,
    const __bf16* __restrict__ BT, int ldb,
    float* __restrict__ Cf, __bf16* __restrict__ Cb, int ldc,
    int M, int N, int K, float alpha, int epi, int wave, int lane)
{
  __syncthreads();
  const int tilesN = N >> 4;
  const int ntiles = (M >> 4) * tilesN;
  const int l  = lane & 15;
  const bool hi = lane >= 16;
  const int aoff = hi ? 8 : 0;
  const int boff = hi ? 8 : 0;
  for (int t = wave; t < ntiles; t += NWAVES) {
    const int row0 = (t / tilesN) << 4;
    const int col0 = (t % tilesN) << 4;
    v8f acc = {};
    for (int k0 = 0; k0 < K; k0 += 32) {
      const __bf16* ap = A  + (row0 + l) * lda + k0 + aoff;
      const __bf16* bp = BT + (col0 + l) * ldb + k0 + boff;
      BfPack a, b;
      a.u[0] = *(const uint4*)(ap);        // K  k0+aoff   .. +7
      a.u[1] = *(const uint4*)(ap + 16);   // K  k0+16+aoff.. +7
      b.u[0] = *(const uint4*)(bp);        // K  k0+boff   .. +7
      b.u[1] = *(const uint4*)(bp + 16);   // K  k0+16+boff.. +7
      acc = __builtin_amdgcn_wmma_f32_16x16x32_bf16(false, a.v, false, b.v,
                                                    (short)0, acc, false, false);
    }
    const int ro = hi ? 8 : 0;
#pragma unroll
    for (int r = 0; r < 8; ++r) {
      const int rr = row0 + r + ro, cc = col0 + l;
      const float v = alpha * acc[r];
      if      (epi == EP_F32)  Cf[rr * ldc + cc] = v;
      else if (epi == EP_BF)   Cb[rr * ldc + cc] = to_bf(v);
      else if (epi == EP_BF_T) Cb[cc * ldc + rr] = to_bf(v);
      else                     Cb[rr * ldc + cc] = to_bf(((rr == cc) ? 3.f : 0.f) - acc[r]);
    }
  }
  __syncthreads();
}

__device__ __forceinline__ float block_sum(float v, float* red, int tid)
{
  red[tid] = v;
  __syncthreads();
  for (int off = NTHREADS / 2; off > 0; off >>= 1) {
    if (tid < off) red[tid] += red[tid + off];
    __syncthreads();
  }
  float r = red[0];
  __syncthreads();
  return r;
}

// ---------------------------------------------------------------------------
// ReEig (bf16 Newton-Schulz sign, f32 master):
//   S <- 0.5*(S + eps I + sign(S-eps I)*(S-eps I))
// sign via Y <- 0.5*Y*(3I - Y^2) in bf16/f32-accum; all operands symmetric.
// On exit S (f32) and Sb (bf16 mirror, = next layer's A operand) hold result.
// iters must be even.
// ---------------------------------------------------------------------------
__device__ __forceinline__ void reeig_bf(
    float* S, int ldS, float* P,
    __bf16* Sb, __bf16* Y1, __bf16* Y2, __bf16* Hb, int ldb,
    int n, int iters, float* red, int tid, int wave, int lane)
{
  float s = 0.f;
  for (int idx = tid; idx < n * n; idx += NTHREADS) {
    int i = idx / n, j = idx - i * n;
    float v = S[i * ldS + j];
    Sb[i * ldb + j] = to_bf(v);
    v -= (i == j) ? SPD_EPS : 0.f;
    s += v * v;
  }
  float c   = sqrtf(block_sum(s, red, tid)) + 1e-20f;
  float inv = 1.f / c;
  for (int idx = tid; idx < n * n; idx += NTHREADS) {
    int i = idx / n, j = idx - i * n;
    Y1[i * ldb + j] = to_bf((S[i * ldS + j] - ((i == j) ? SPD_EPS : 0.f)) * inv);
  }
  __bf16* Ya = Y1;
  __bf16* Yb = Y2;
  for (int it = 0; it < iters; ++it) {
    gemm_bf(Ya, ldb, Ya, ldb, nullptr, Hb, ldb, n, n, n, 1.f, EP_BF_H, wave, lane); // H=3I-Y^2
    gemm_bf(Ya, ldb, Hb, ldb, nullptr, Yb, ldb, n, n, n, 0.5f, EP_BF, wave, lane);  // Y=0.5*Y*H
    __bf16* t = Ya; Ya = Yb; Yb = t;
  }
  // P = sign * S   (sign*M = sign*S - eps*sign, folded into combine below)
  gemm_bf(Ya, ldb, Sb, ldb, P, nullptr, ldS, n, n, n, 1.f, EP_F32, wave, lane);
  for (int idx = tid; idx < n * n; idx += NTHREADS) {
    int i = idx / n, j = idx - i * n;
    float ns = 0.5f * (S[i * ldS + j] + P[i * ldS + j] - SPD_EPS * from_bf(Ya[i * ldb + j]))
             + ((i == j) ? 0.5f * SPD_EPS : 0.f);
    S[i * ldS + j]  = ns;
    Sb[i * ldb + j] = to_bf(ns);
  }
  __syncthreads();
}

// f32 ReEig for the final 16x16 stage (K=16 < 32, accuracy matters most here).
__device__ __forceinline__ void reeig_ns(
    float* S, float* Y0, float* P, float* Ping,
    int n, int ld, int iters, float* red, int tid, int wave, int lane)
{
  float s = 0.f;
  for (int idx = tid; idx < n * n; idx += NTHREADS) {
    int i = idx / n, j = idx - i * n;
    float v = S[i * ld + j] - ((i == j) ? SPD_EPS : 0.f);
    s += v * v;
  }
  float c   = sqrtf(block_sum(s, red, tid)) + 1e-20f;
  float inv = 1.f / c;
  for (int idx = tid; idx < n * n; idx += NTHREADS) {
    int i = idx / n, j = idx - i * n;
    Y0[i * ld + j] = (S[i * ld + j] - ((i == j) ? SPD_EPS : 0.f)) * inv;
  }
  float* Ya = Y0;
  float* Yb = Ping;
  for (int it = 0; it < iters; ++it) {
    gemm_tiles(Ya, ld, 0, Ya, ld, P, ld, n, n, n, 1.f, wave, lane);
    for (int idx = tid; idx < n * n; idx += NTHREADS) {
      int i = idx / n, j = idx - i * n;
      P[i * ld + j] = ((i == j) ? 3.f : 0.f) - P[i * ld + j];
    }
    gemm_tiles(Ya, ld, 0, P, ld, Yb, ld, n, n, n, 0.5f, wave, lane);
    float* t = Ya; Ya = Yb; Yb = t;
  }
  gemm_tiles(Ya, ld, 0, S, ld, P, ld, n, n, n, 1.f, wave, lane);
  for (int idx = tid; idx < n * n; idx += NTHREADS) {
    int i = idx / n, j = idx - i * n;
    S[i * ld + j] = 0.5f * (S[i * ld + j] + P[i * ld + j] - SPD_EPS * Ya[i * ld + j])
                  + ((i == j) ? 0.5f * SPD_EPS : 0.f);
  }
  __syncthreads();
}

// ---------------------------------------------------------------------------
// LogEig on 16x16 SPD (f32): 4 inversion-free coupled-NS square roots,
// then 8-term Mercator series.  log S = 2^4 log((S/c)^(1/16)) + log(c) I.
// ---------------------------------------------------------------------------
__device__ __forceinline__ void logeig16(
    float* S, float* work, float* red, int tid, int wave, int lane)
{
  const int ld = LD16;
  float s = 0.f;
  for (int idx = tid; idx < 256; idx += NTHREADS) {
    int i = idx >> 4, j = idx & 15;
    float v = S[i * ld + j];
    s += v * v;
  }
  float c   = sqrtf(block_sum(s, red, tid)) + 1e-20f;
  float inv = 1.f / c;

  float* Bc = work;
  float* Y  = work + 320;
  float* Z  = work + 640;
  float* P  = work + 960;
  float* Yp = work + 1280;
  float* Zp = work + 1600;

  for (int idx = tid; idx < 256; idx += NTHREADS) {
    int i = idx >> 4, j = idx & 15;
    Bc[i * ld + j] = S[i * ld + j] * inv;
  }
  __syncthreads();

  for (int sq = 0; sq < 4; ++sq) {
    for (int idx = tid; idx < 256; idx += NTHREADS) {
      int i = idx >> 4, j = idx & 15;
      Y[i * ld + j] = Bc[i * ld + j];
      Z[i * ld + j] = (i == j) ? 1.f : 0.f;
    }
    float *Ya = Y, *Yb = Yp, *Za = Z, *Zb = Zp;
    for (int it = 0; it < 12; ++it) {
      gemm_tiles(Za, ld, 0, Ya, ld, P, ld, 16, 16, 16, 1.f, wave, lane);
      for (int idx = tid; idx < 256; idx += NTHREADS) {
        int i = idx >> 4, j = idx & 15;
        P[i * ld + j] = ((i == j) ? 3.f : 0.f) - P[i * ld + j];
      }
      gemm_tiles(Ya, ld, 0, P, ld, Yb, ld, 16, 16, 16, 0.5f, wave, lane);
      gemm_tiles(P, ld, 0, Za, ld, Zb, ld, 16, 16, 16, 0.5f, wave, lane);
      float* t = Ya; Ya = Yb; Yb = t;
      t = Za; Za = Zb; Zb = t;
    }
    for (int idx = tid; idx < 256; idx += NTHREADS) {
      int i = idx >> 4, j = idx & 15;
      Bc[i * ld + j] = Ya[i * ld + j];
    }
    __syncthreads();
  }

  for (int idx = tid; idx < 256; idx += NTHREADS) {
    int i = idx >> 4, j = idx & 15;
    Y[i * ld + j] = Bc[i * ld + j] - ((i == j) ? 1.f : 0.f);   // E
  }
  for (int idx = tid; idx < 256; idx += NTHREADS) {
    int i = idx >> 4, j = idx & 15;
    Z[i * ld + j] = (-1.f / 8.f) * Y[i * ld + j] + ((i == j) ? (1.f / 7.f) : 0.f);
  }
  float* Qa = Z;
  float* Qb = Zp;
  for (int k = 6; k >= 1; --k) {
    gemm_tiles(Qa, ld, 0, Y, ld, Qb, ld, 16, 16, 16, 1.f, wave, lane);
    float ck = ((k & 1) ? 1.f : -1.f) / (float)k;
    for (int idx = tid; idx < 256; idx += NTHREADS) {
      int i = idx >> 4, j = idx & 15;
      if (i == j) Qb[i * ld + j] += ck;
    }
    float* t = Qa; Qa = Qb; Qb = t;
  }
  gemm_tiles(Y, ld, 0, Qa, ld, P, ld, 16, 16, 16, 1.f, wave, lane);   // L = E*Q
  float lc = logf(c);
  for (int idx = tid; idx < 256; idx += NTHREADS) {
    int i = idx >> 4, j = idx & 15;
    S[i * ld + j] = 16.f * P[i * ld + j] + ((i == j) ? lc : 0.f);
  }
  __syncthreads();
}

// ---------------------------------------------------------------------------
// One workgroup (8 wave32) per batch element; everything stays in LDS.
// bf16 arena (reused per stage):
//   L1:  Xbf[0,17408)  W1T[17408,26112)  T1T[26112,34816)
//   NS64: Sb[0) Y1[4608) Y2[9216) H[13824)        (ld 72)
//   L2:  H1b=Sb[0)  W2T[4608)  T2T[6912)
//   NS32: Sb[0) Y1[1280) Y2[2560) H[3840)          (ld 40)
//   L3:  H2b=Sb[0)  W3T[1280)  T3T[1920)
// ---------------------------------------------------------------------------
__global__ __launch_bounds__(NTHREADS) void spdnet_kernel(
    const float* __restrict__ x,  const float* __restrict__ W1,
    const float* __restrict__ W2, const float* __restrict__ W3,
    const float* __restrict__ fcw, const float* __restrict__ fcb,
    float* __restrict__ out)
{
  __shared__ float pool[2 * 64 * LD64];                     // 8704 f32: S / P
  __shared__ float red[NTHREADS];
  __shared__ __attribute__((aligned(16))) __bf16 arena[34816];

  const int tid  = threadIdx.x;
  // Scalar-uniform wave id: lets the backend use s_cbranch for tile loops and
  // keeps EXEC untouched (all-ones) around every WMMA.
  const int wave = __builtin_amdgcn_readfirstlane(tid >> 5);
  const int lane = tid & 31;
  const int b    = blockIdx.x;
  const float* Xg = x + (size_t)b * (128 * 128);
  float* Pbuf = pool + 64 * LD64;

  // ---- layer 1: S1 = W1^T X W1 (128 -> 64) --------------------------------
  __bf16* Xbf = arena;            // 128 x 136
  __bf16* W1T = arena + 17408;    // 64 x 136  (= W1^T)
  __bf16* T1T = arena + 26112;    // 64 x 136  (= (X W1)^T)
  for (int i = tid; i < 128 * 128; i += NTHREADS) {
    int r = i >> 7, c = i & 127;
    Xbf[r * 136 + c] = to_bf(Xg[i]);
  }
  for (int i = tid; i < 128 * 64; i += NTHREADS) {
    int k = i >> 6, m = i & 63;
    W1T[m * 136 + k] = to_bf(W1[i]);
  }
  gemm_bf(Xbf, 136, W1T, 136, nullptr, T1T, 136, 128, 64, 128, 1.f, EP_BF_T, wave, lane);
  gemm_bf(W1T, 136, T1T, 136, pool, nullptr, LD64, 64, 64, 128, 1.f, EP_F32, wave, lane);
  reeig_bf(pool, LD64, Pbuf, arena, arena + 4608, arena + 9216, arena + 13824,
           72, 64, 12, red, tid, wave, lane);

  // ---- layer 2: S2 = W2^T H1 W2 (64 -> 32) --------------------------------
  __bf16* H1b = arena;            // 64 x 72 (bf16 mirror from reeig_bf)
  __bf16* W2T = arena + 4608;     // 32 x 72
  __bf16* T2T = arena + 6912;     // 32 x 72
  for (int i = tid; i < 64 * 32; i += NTHREADS) {
    int k = i >> 5, m = i & 31;
    W2T[m * 72 + k] = to_bf(W2[i]);
  }
  gemm_bf(H1b, 72, W2T, 72, nullptr, T2T, 72, 64, 32, 64, 1.f, EP_BF_T, wave, lane);
  gemm_bf(W2T, 72, T2T, 72, pool, nullptr, LD32, 32, 32, 64, 1.f, EP_F32, wave, lane);
  reeig_bf(pool, LD32, Pbuf, arena, arena + 1280, arena + 2560, arena + 3840,
           40, 32, 10, red, tid, wave, lane);

  // ---- layer 3: S3 = W3^T H2 W3 (32 -> 16) --------------------------------
  __bf16* H2b = arena;            // 32 x 40
  __bf16* W3T = arena + 1280;     // 16 x 40
  __bf16* T3T = arena + 1920;     // 16 x 40
  for (int i = tid; i < 32 * 16; i += NTHREADS) {
    int k = i >> 4, m = i & 15;
    W3T[m * 40 + k] = to_bf(W3[i]);
  }
  gemm_bf(H2b, 40, W3T, 40, nullptr, T3T, 40, 32, 16, 32, 1.f, EP_BF_T, wave, lane);
  gemm_bf(W3T, 40, T3T, 40, pool, nullptr, LD16, 16, 16, 32, 1.f, EP_F32, wave, lane);

  // ---- final 16x16: ReEig + LogEig in f32 WMMA ----------------------------
  reeig_ns(pool, pool + 512, pool + 1024, pool + 1536, 16, LD16, 8, red, tid, wave, lane);
  logeig16(pool, pool + 2048, red, tid, wave, lane);

  // ---- vech (row-major upper triangle) + FC -------------------------------
  __syncthreads();
  if (tid < 136) {
    int i = 0, t = tid;
    while (t >= 16 - i) { t -= 16 - i; ++i; }
    int j = i + t;
    red[tid] = pool[i * LD16 + j];
  }
  __syncthreads();
  if (tid < 7) {
    float acc = fcb[tid];
    for (int k = 0; k < 136; ++k) acc += red[k] * fcw[tid * 136 + k];
    out[(size_t)b * 7 + tid] = acc;
  }
}

extern "C" void kernel_launch(void* const* d_in, const int* in_sizes, int n_in,
                              void* d_out, int out_size, void* d_ws, size_t ws_size,
                              hipStream_t stream)
{
  const float* x   = (const float*)d_in[0];
  const float* W1  = (const float*)d_in[1];
  const float* W2  = (const float*)d_in[2];
  const float* W3  = (const float*)d_in[3];
  const float* fcw = (const float*)d_in[4];
  const float* fcb = (const float*)d_in[5];
  float* out = (float*)d_out;
  const int B = in_sizes[0] / (128 * 128);   // 8192
  spdnet_kernel<<<B, NTHREADS, 0, stream>>>(x, W1, W2, W3, fcw, fcb, out);
}
